// TokenPerformer_59330678227088
// MI455X (gfx1250) — matrix operand order repaired and checked by reading, not compiled, not run on hardware
//
#include <hip/hip_runtime.h>
#include <math.h>

// ---------------------------------------------------------------------------
// Token Performer block for MI455X (gfx1250, wave32, WMMA)
//   B=32, T=3136, DIM=EMB=512, M=256, NT = B*T = 100352 tokens
// All GEMMs run on v_wmma_f32_16x16x32_f16 (f16 operands, f32 accum).
// Register-blocked: each wave computes a 32x64 macro-tile (2x4 WMMA tiles),
// reusing A fragments 4x and B fragments 2x -> 1.5 b128 loads per WMMA.
// GEMM is templated on epilogue MODE so each instantiation carries only its
// own (spill-free) epilogue; batch math is hoisted to wave-uniform scalars.
// ---------------------------------------------------------------------------

typedef __attribute__((ext_vector_type(16))) _Float16 v16h;
typedef __attribute__((ext_vector_type(8)))  _Float16 v8h;
typedef __attribute__((ext_vector_type(8)))  float    v8f;

#define BATCH 32
#define TOK   3136
#define NTOK  (BATCH * TOK)   // 100352

// ---------------------------------------------------------------------------
// Small helper kernels
// ---------------------------------------------------------------------------

// transpose-convert: src f32 [K][N] row-major -> dst f16 [N][K]
__global__ void k_txp16(const float* __restrict__ src, _Float16* __restrict__ dst,
                        int K, int N) {
  long long idx = (long long)blockIdx.x * blockDim.x + threadIdx.x;
  if (idx < (long long)K * N) {
    int k = (int)(idx / N);
    int n = (int)(idx % N);
    dst[(long long)n * K + k] = (_Float16)src[idx];
  }
}

// straight convert f32 -> f16
__global__ void k_cvt16(const float* __restrict__ src, _Float16* __restrict__ dst,
                        long long n) {
  long long idx = (long long)blockIdx.x * blockDim.x + threadIdx.x;
  if (idx < n) dst[idx] = (_Float16)src[idx];
}

__global__ void k_zero(float* __restrict__ p, int n) {
  int idx = blockIdx.x * blockDim.x + threadIdx.x;
  if (idx < n) p[idx] = 0.0f;
}

// LayerNorm over width 512: one block (256 thr) per token, f32 in -> f16 out
__global__ void k_ln512(const float* __restrict__ x, const float* __restrict__ g,
                        const float* __restrict__ b, _Float16* __restrict__ out) {
  __shared__ float s1[256], s2[256];
  int tid = threadIdx.x;
  long long t = blockIdx.x;
  const float* xr = x + t * 512;
  float a0 = xr[tid], a1 = xr[tid + 256];
  s1[tid] = a0 + a1;
  s2[tid] = a0 * a0 + a1 * a1;
  __syncthreads();
  for (int off = 128; off > 0; off >>= 1) {
    if (tid < off) { s1[tid] += s1[tid + off]; s2[tid] += s2[tid + off]; }
    __syncthreads();
  }
  float mu = s1[0] * (1.0f / 512.0f);
  float var = s2[0] * (1.0f / 512.0f) - mu * mu;
  float rs = rsqrtf(var + 1e-3f);
  out[t * 512 + tid]       = (_Float16)((a0 - mu) * rs * g[tid] + b[tid]);
  out[t * 512 + tid + 256] = (_Float16)((a1 - mu) * rs * g[tid + 256] + b[tid + 256]);
}

// xd[t] = 0.5 * sum_i src[t][i]^2  over width 512 (f16 src)
__global__ void k_rowsq512(const _Float16* __restrict__ src, float* __restrict__ xd) {
  __shared__ float s1[256];
  int tid = threadIdx.x;
  long long t = blockIdx.x;
  float a0 = (float)src[t * 512 + tid];
  float a1 = (float)src[t * 512 + tid + 256];
  s1[tid] = a0 * a0 + a1 * a1;
  __syncthreads();
  for (int off = 128; off > 0; off >>= 1) {
    if (tid < off) s1[tid] += s1[tid + off];
    __syncthreads();
  }
  if (tid == 0) xd[t] = s1[0] * 0.5f;
}

// den[t] = sum_m qp[t][m] * kpsum[batch(t)][m]   (M = 256 = blockDim)
__global__ void k_dden(const _Float16* __restrict__ qp, const float* __restrict__ kpsum,
                       float* __restrict__ den) {
  __shared__ float s1[256];
  int tid = threadIdx.x;
  long long t = blockIdx.x;
  int b = (int)(t / TOK);
  s1[tid] = (float)qp[t * 256 + tid] * kpsum[(long long)b * 256 + tid];
  __syncthreads();
  for (int off = 128; off > 0; off >>= 1) {
    if (tid < off) s1[tid] += s1[tid + off];
    __syncthreads();
  }
  if (tid == 0) den[t] = s1[0];
}

// ---------------------------------------------------------------------------
// Register-blocked WMMA GEMM, templated on epilogue MODE.
//   A[rows,K] row-major f16, Bt[N][K] f16, f32 accumulation.
//   Block = 128 thr (4 waves) -> block tile 64 rows x 128 cols.
//   Wave tile = 32 rows x 64 cols = 2x4 WMMA 16x16 tiles.
// ---------------------------------------------------------------------------
struct GemmArgs {
  const _Float16* A;
  const _Float16* Bt;
  long long lda, ldbt;       // element strides of A rows / Bt rows
  long long aBatch, bBatch;  // per-blockIdx.z (or per-row-batch) element strides
  int K;
  const float* bias;
  const float* xd;           // mode 1/2: -|x|^2/2 term
  const float* den;          // mode 4: attention denominator
  const float* add32;        // mode 5/7: residual source
  float*     o32;
  _Float16*  o16;
  _Float16*  o16b;           // secondary f16 output (q in mode 0)
  _Float16*  o16c;           // tertiary f16 output (vT in mode 0)
  float*     sums;           // mode 1: kpsum atomic accumulation
};

template <int MODE>
__global__ __launch_bounds__(128) void k_gemm(GemmArgs p) {
  const int lane  = threadIdx.x & 31;
  const int wave  = threadIdx.x >> 5;
  const int waveM = wave & 1;
  const int waveN = wave >> 1;
  const int row0 = blockIdx.y * 64 + waveM * 32;   // wave's first row
  const int col0 = blockIdx.x * 128 + waveN * 64;  // wave's first col
  const int m  = lane & 15;   // A row within tile == D column (N) within tile
  const int hi = lane >> 4;

  const _Float16* Abase  = p.A  + (long long)blockIdx.z * p.aBatch;
  const _Float16* Btbase = p.Bt + (long long)blockIdx.z * p.bBatch;
  if (MODE == 4) Btbase = p.Bt + (long long)(row0 / TOK) * p.bBatch;

  const _Float16* arow[2];
#pragma unroll
  for (int mi = 0; mi < 2; ++mi)
    arow[mi] = Abase + (long long)(row0 + mi * 16 + m) * p.lda;
  const _Float16* brow[4];
#pragma unroll
  for (int ni = 0; ni < 4; ++ni)
    brow[ni] = Btbase + (long long)(col0 + ni * 16 + m) * p.ldbt;

  v8f acc[2][4];
#pragma unroll
  for (int mi = 0; mi < 2; ++mi)
#pragma unroll
    for (int ni = 0; ni < 4; ++ni) acc[mi][ni] = (v8f){};

  for (int k0 = 0; k0 < p.K; k0 += 32) {
    v16h af[2], bf[4];
#pragma unroll
    for (int mi = 0; mi < 2; ++mi) {
      // A 16x32 frag: halves 0..7 -> K=hi*8..+7 ; 8..15 -> K=16+hi*8..+7
      v8h alo = *(const v8h*)(arow[mi] + k0 + hi * 8);
      v8h ahi = *(const v8h*)(arow[mi] + k0 + 16 + hi * 8);
#pragma unroll
      for (int j = 0; j < 8; ++j) { af[mi][j] = alo[j]; af[mi][j + 8] = ahi[j]; }
    }
#pragma unroll
    for (int ni = 0; ni < 4; ++ni) {
      // B 32x16 frag (from Bt[N][K]): halves 0..7 -> K=hi*8..+7 ; 8..15 -> K=16+hi*8..+7
      v8h blo = *(const v8h*)(brow[ni] + k0 + hi * 8);
      v8h bhi = *(const v8h*)(brow[ni] + k0 + 16 + hi * 8);
#pragma unroll
      for (int j = 0; j < 8; ++j) { bf[ni][j] = blo[j]; bf[ni][j + 8] = bhi[j]; }
    }
#pragma unroll
    for (int mi = 0; mi < 2; ++mi)
#pragma unroll
      for (int ni = 0; ni < 4; ++ni)
        acc[mi][ni] = __builtin_amdgcn_wmma_f32_16x16x32_f16(
            false, af[mi], false, bf[ni], (short)0, acc[mi][ni], false, false);
  }

  // ---- epilogue ------------------------------------------------------------
  // Wave tile (32 rows) never crosses a batch boundary (TOK = 98*32).
  const int bat = row0 / TOK;          // wave-uniform
  const int t0  = row0 - bat * TOK;    // wave-uniform

#pragma unroll
  for (int mi = 0; mi < 2; ++mi) {
    const int R  = row0 + mi * 16 + hi * 8;  // first of this lane's 8 rows
    const int tR = t0 + mi * 16 + hi * 8;    // same, within batch
#pragma unroll
    for (int ni = 0; ni < 4; ++ni) {
      const int col = col0 + ni * 16 + m;
      const v8f a = acc[mi][ni];

      if (MODE == 0) {  // kqv -> k16 / q16 / (v32 + packed vT16)
        const float bia = p.bias[col];
        if (col < 512) {
          _Float16* o = p.o16 + (long long)R * 512 + col;
#pragma unroll
          for (int r = 0; r < 8; ++r) o[r * 512] = (_Float16)(a[r] + bia);
        } else if (col < 1024) {
          _Float16* o = p.o16b + (long long)R * 512 + (col - 512);
#pragma unroll
          for (int r = 0; r < 8; ++r) o[r * 512] = (_Float16)(a[r] + bia);
        } else {
          const int cc = col - 1024;
          float* ov = p.o32 + (long long)R * 512 + cc;
          v8h pack;
#pragma unroll
          for (int r = 0; r < 8; ++r) {
            float e = a[r] + bia;
            ov[r * 512] = e;
            pack[r] = (_Float16)e;
          }
          *(v8h*)(p.o16c + ((long long)bat * 512 + cc) * TOK + tR) = pack;  // v^T
        }
      } else if (MODE == 1) {  // kp, stored transposed [b][m][t] + kpsum
        float partial = 0.0f;
        v8h pack;
#pragma unroll
        for (int r = 0; r < 8; ++r) {
          float e = expf(a[r] - p.xd[R + r]) * 0.0625f;
          pack[r] = (_Float16)e;
          partial += e;
        }
        *(v8h*)(p.o16 + ((long long)bat * 256 + col) * TOK + tR) = pack;
        atomicAdd(&p.sums[bat * 256 + col], partial);
      } else if (MODE == 2) {  // qp, natural layout [t][m]
        _Float16* o = p.o16 + (long long)R * 256 + col;
#pragma unroll
        for (int r = 0; r < 8; ++r)
          o[r * 256] = (_Float16)(expf(a[r] - p.xd[R + r]) * 0.0625f);
      } else if (MODE == 3) {  // kptv[b][n][m]
        _Float16* o = p.o16 + ((long long)blockIdx.z * 512 + R) * 256 + col;
#pragma unroll
        for (int r = 0; r < 8; ++r) o[r * 256] = (_Float16)a[r];
      } else if (MODE == 4) {  // attn = acc / (D + eps)
        _Float16* o = p.o16 + (long long)R * 512 + col;
#pragma unroll
        for (int r = 0; r < 8; ++r)
          o[r * 512] = (_Float16)(a[r] / (p.den[R + r] + 1e-8f));
      } else if (MODE == 5) {  // y = v + (attn @ proj_w + proj_b)
        const float bia = p.bias[col];
        const float* rv = p.add32 + (long long)R * 512 + col;
        float* o = p.o32 + (long long)R * 512 + col;
#pragma unroll
        for (int r = 0; r < 8; ++r) o[r * 512] = a[r] + bia + rv[r * 512];
      } else if (MODE == 6) {  // gelu(h2 @ w1 + b1), exact erf form
        const float bia = p.bias[col];
        _Float16* o = p.o16 + (long long)R * 512 + col;
#pragma unroll
        for (int r = 0; r < 8; ++r) {
          float e = a[r] + bia;
          e = 0.5f * e * (1.0f + erff(e * 0.70710678118654752f));
          o[r * 512] = (_Float16)e;
        }
      } else if (MODE == 7) {  // out = y + (g @ w2 + b2)
        const float bia = p.bias[col];
        const float* ry = p.add32 + (long long)R * 512 + col;
        float* o = p.o32 + (long long)R * 512 + col;
#pragma unroll
        for (int r = 0; r < 8; ++r) o[r * 512] = a[r] + bia + ry[r * 512];
      }
    }
  }
}

// ---------------------------------------------------------------------------
// Host-side orchestration
// ---------------------------------------------------------------------------
extern "C" void kernel_launch(void* const* d_in, const int* in_sizes, int n_in,
                              void* d_out, int out_size, void* d_ws, size_t ws_size,
                              hipStream_t stream) {
  const float* x      = (const float*)d_in[0];
  const float* kqv_w  = (const float*)d_in[1];
  const float* kqv_b  = (const float*)d_in[2];
  const float* proj_w = (const float*)d_in[3];
  const float* proj_b = (const float*)d_in[4];
  const float* n1_g   = (const float*)d_in[5];
  const float* n1_b   = (const float*)d_in[6];
  const float* n2_g   = (const float*)d_in[7];
  const float* n2_b   = (const float*)d_in[8];
  const float* mlp_w1 = (const float*)d_in[9];
  const float* mlp_b1 = (const float*)d_in[10];
  const float* mlp_w2 = (const float*)d_in[11];
  const float* mlp_b2 = (const float*)d_in[12];
  const float* wrf    = (const float*)d_in[13];
  float* out = (float*)d_out;

  // ---- workspace carve-up (256B aligned) ----
  size_t off = 0;
  auto alloc = [&](size_t bytes) -> char* {
    char* p = (char*)d_ws + off;
    off += (bytes + 255) & ~(size_t)255;
    return p;
  };
  const size_t NT = NTOK;
  _Float16* h16     = (_Float16*)alloc(NT * 512 * 2);
  _Float16* k16     = (_Float16*)alloc(NT * 512 * 2);
  _Float16* q16     = (_Float16*)alloc(NT * 512 * 2);
  float*    v32     = (float*)   alloc(NT * 512 * 4);
  _Float16* vT16    = (_Float16*)alloc(NT * 512 * 2);
  _Float16* kpT16   = (_Float16*)alloc(NT * 256 * 2);
  _Float16* qp16    = (_Float16*)alloc(NT * 256 * 2);
  float*    kpsum   = (float*)   alloc((size_t)BATCH * 256 * 4);
  float*    xdk     = (float*)   alloc(NT * 4);
  float*    xdq     = (float*)   alloc(NT * 4);
  float*    dden    = (float*)   alloc(NT * 4);
  _Float16* kptv16  = (_Float16*)alloc((size_t)BATCH * 512 * 256 * 2);
  _Float16* attn16  = (_Float16*)alloc(NT * 512 * 2);
  float*    y32     = (float*)   alloc(NT * 512 * 4);
  _Float16* h2_16   = (_Float16*)alloc(NT * 512 * 2);
  _Float16* g16     = (_Float16*)alloc(NT * 512 * 2);
  _Float16* kqvwT   = (_Float16*)alloc((size_t)1536 * 512 * 2);
  _Float16* projwT  = (_Float16*)alloc((size_t)512 * 512 * 2);
  _Float16* w1T     = (_Float16*)alloc((size_t)512 * 512 * 2);
  _Float16* w2T     = (_Float16*)alloc((size_t)512 * 512 * 2);
  _Float16* wrf16   = (_Float16*)alloc((size_t)256 * 512 * 2);

  const int rowBlocks = (int)(NT / 64);  // 1568

  // ---- weight preparation: Bt[N][K] f16 ----
  k_txp16<<<(512 * 1536 + 255) / 256, 256, 0, stream>>>(kqv_w, kqvwT, 512, 1536);
  k_txp16<<<(512 * 512 + 255) / 256, 256, 0, stream>>>(proj_w, projwT, 512, 512);
  k_txp16<<<(512 * 512 + 255) / 256, 256, 0, stream>>>(mlp_w1, w1T, 512, 512);
  k_txp16<<<(512 * 512 + 255) / 256, 256, 0, stream>>>(mlp_w2, w2T, 512, 512);
  k_cvt16<<<(256 * 512 + 255) / 256, 256, 0, stream>>>(wrf, wrf16, 256 * 512);

  // ---- norm1 ----
  k_ln512<<<NT, 256, 0, stream>>>(x, n1_g, n1_b, h16);

  // ---- kqv GEMM (NT x 512 x 1536) ----
  {
    GemmArgs g = {};
    g.A = h16; g.Bt = kqvwT; g.lda = 512; g.ldbt = 512; g.K = 512;
    g.bias = kqv_b;
    g.o16 = k16; g.o16b = q16; g.o32 = v32; g.o16c = vT16;
    k_gemm<0><<<dim3(1536 / 128, rowBlocks, 1), 128, 0, stream>>>(g);
  }

  // ---- |k|^2/2, |q|^2/2 ----
  k_rowsq512<<<NT, 256, 0, stream>>>(k16, xdk);
  k_rowsq512<<<NT, 256, 0, stream>>>(q16, xdq);
  k_zero<<<(BATCH * 256 + 255) / 256, 256, 0, stream>>>(kpsum, BATCH * 256);

  // ---- kp = prm_exp(k) stored [b][m][t] + kpsum via f32 atomics ----
  {
    GemmArgs g = {};
    g.A = k16; g.Bt = wrf16; g.lda = 512; g.ldbt = 512; g.K = 512;
    g.xd = xdk; g.o16 = kpT16; g.sums = kpsum;
    k_gemm<1><<<dim3(256 / 128, rowBlocks, 1), 128, 0, stream>>>(g);
  }
  // ---- qp = prm_exp(q) stored [t][m] ----
  {
    GemmArgs g = {};
    g.A = q16; g.Bt = wrf16; g.lda = 512; g.ldbt = 512; g.K = 512;
    g.xd = xdq; g.o16 = qp16;
    k_gemm<2><<<dim3(256 / 128, rowBlocks, 1), 128, 0, stream>>>(g);
  }

  // ---- D[t] = qp . kpsum ----
  k_dden<<<NT, 256, 0, stream>>>(qp16, kpsum, dden);

  // ---- kptv[b][n][m] = sum_t v[t,n] kp[t,m]  (per-batch 512x3136x256) ----
  {
    GemmArgs g = {};
    g.A = vT16; g.Bt = kpT16; g.lda = TOK; g.ldbt = TOK;
    g.aBatch = (long long)512 * TOK; g.bBatch = (long long)256 * TOK;
    g.K = TOK; g.o16 = kptv16;
    k_gemm<3><<<dim3(256 / 128, 512 / 64, BATCH), 128, 0, stream>>>(g);
  }

  // ---- attn = (qp @ kptv^T) / (D + eps) ----
  {
    GemmArgs g = {};
    g.A = qp16; g.Bt = kptv16; g.lda = 256; g.ldbt = 256;
    g.bBatch = (long long)512 * 256;
    g.K = 256; g.den = dden; g.o16 = attn16;
    k_gemm<4><<<dim3(512 / 128, rowBlocks, 1), 128, 0, stream>>>(g);
  }

  // ---- y = v + attn @ proj_w + proj_b ----
  {
    GemmArgs g = {};
    g.A = attn16; g.Bt = projwT; g.lda = 512; g.ldbt = 512; g.K = 512;
    g.bias = proj_b; g.add32 = v32; g.o32 = y32;
    k_gemm<5><<<dim3(512 / 128, rowBlocks, 1), 128, 0, stream>>>(g);
  }

  // ---- norm2 + MLP + residual ----
  k_ln512<<<NT, 256, 0, stream>>>(y32, n2_g, n2_b, h2_16);
  {
    GemmArgs g = {};
    g.A = h2_16; g.Bt = w1T; g.lda = 512; g.ldbt = 512; g.K = 512;
    g.bias = mlp_b1; g.o16 = g16;
    k_gemm<6><<<dim3(512 / 128, rowBlocks, 1), 128, 0, stream>>>(g);
  }
  {
    GemmArgs g = {};
    g.A = g16; g.Bt = w2T; g.lda = 512; g.ldbt = 512; g.K = 512;
    g.bias = mlp_b2; g.add32 = y32; g.o32 = out;
    k_gemm<7><<<dim3(512 / 128, rowBlocks, 1), 128, 0, stream>>>(g);
  }
}
